// MultiHeadAttention_6442450944465
// MI455X (gfx1250) — matrix operand order repaired and checked by reading, not compiled, not run on hardware
//
#include <hip/hip_runtime.h>
#include <hip/hip_bf16.h>

// ---------------------------------------------------------------------------
// MHA forward for MI455X (gfx1250, wave32, WMMA 16x16x32 bf16 + TDM).
// B=4, T=2048, C=1024, H=16, D=64.
// ---------------------------------------------------------------------------

#define B_  4
#define T_  2048
#define C_  1024
#define H_  16
#define D_  64
#define BT_ (B_ * T_)   // 8192

typedef __attribute__((ext_vector_type(16))) __bf16 v16bf;
typedef __attribute__((ext_vector_type(8)))  float  v8f;
typedef __attribute__((ext_vector_type(4)))  unsigned int v4u;
typedef __attribute__((ext_vector_type(8)))  int    v8i;
typedef __attribute__((ext_vector_type(4)))  int    v4i;

union Frag16 {          // 16 bf16 = one WMMA A/B fragment per lane
    uint4  u[2];
    v16bf  v;
};

__device__ __forceinline__ v8f v8f_zero() {
    v8f z = {0.f, 0.f, 0.f, 0.f, 0.f, 0.f, 0.f, 0.f};
    return z;
}

__device__ __forceinline__ v8f wmma_bf16(const v16bf& a, const v16bf& b, const v8f& c) {
    return __builtin_amdgcn_wmma_f32_16x16x32_bf16(
        /*neg_a=*/false, a, /*neg_b=*/false, b,
        /*c_mod=*/(short)0, c, /*reuse_a=*/false, /*reuse_b=*/false);
}

// Wait until <=1 TDM op in flight. The LDS pointer operand makes the target
// buffer's address ESCAPE into the asm (which has a memory clobber), so the
// compiler must treat the TDM-written tile as modified memory and emit real
// ds_load instructions after the wait (instead of folding reads of a
// "never-stored" LDS global to undef).
__device__ __forceinline__ void tdm_wait_1(const void* lds_ptr) {
    asm volatile("s_wait_tensorcnt 0x1" :: "v"(lds_ptr) : "memory");
}

// ---------------------------------------------------------------------------
// TDM: async-load a 64-row x 32-bf16 (64x32x2 = 4KB) tile into LDS.
//   gaddr            : global address of tile start (row-major, bf16)
//   lds_off_bytes    : destination offset within the block's LDS allocation
//   row_stride_dw    : tensor row stride in DWORDs (bf16 count / 2)
// D# packing per CDNA5 ISA (2D tile, groups 2/3
// zero). data_size=2 (4-byte units), tile_dim0=16 dw, tile_dim1=64 rows.
// This toolchain exposes the 6-arg builtin (extra int32x8 group before cpol).
// ---------------------------------------------------------------------------
__device__ __forceinline__ void tdm_load_64x32(const __bf16* gaddr,
                                               unsigned lds_off_bytes,
                                               unsigned row_stride_dw) {
    unsigned long long ga = (unsigned long long)(uintptr_t)gaddr;
    v4u g0;
    g0.x = 1u;                                              // count=1 (valid)
    g0.y = lds_off_bytes;                                   // lds_addr
    g0.z = (unsigned)(ga & 0xFFFFFFFFu);                    // global_addr[31:0]
    g0.w = (unsigned)((ga >> 32) & 0x01FFFFFFu) | (2u << 30); // addr[56:32]|type=2
    v8i g1;
    g1[0] = (int)(2u << 16);          // workgroup_mask=0, data_size=2 (4B)
    g1[1] = (int)(16u << 16);         // tensor_dim0 = 16 dwords (tile width)
    g1[2] = (int)(0xFFFFu << 16);     // tensor_dim1 lo16 (rows avail, generous)
    g1[3] = (int)(16u << 16);         // tile_dim0 = 16 dwords
    g1[4] = 64;                       // tile_dim1 = 64 rows, tile_dim2 = 0
    g1[5] = (int)row_stride_dw;       // tensor_dim0_stride[31:0]
    g1[6] = 0;                        // stride hi / dim1_stride lo
    g1[7] = 0;
    v4i gz4 = {0, 0, 0, 0};
    v8i gz8 = {0, 0, 0, 0, 0, 0, 0, 0};
    __builtin_amdgcn_tensor_load_to_lds(g0, g1, gz4, gz4, gz8, 0);
}

// ---------------------------------------------------------------------------
// fp32 -> bf16 conversion (grid-stride)
// ---------------------------------------------------------------------------
__global__ void cvt_bf16_kernel(const float* __restrict__ in, __bf16* __restrict__ out, int n) {
    int stride = gridDim.x * blockDim.x;
    for (int i = blockIdx.x * blockDim.x + threadIdx.x; i < n; i += stride)
        out[i] = (__bf16)in[i];
}

// fp32 [nmat, rows, cols] -> bf16 [nmat, cols, rows] (transpose inner 2 dims)
__global__ void cvt_transpose_kernel(const float* __restrict__ in, __bf16* __restrict__ out,
                                     int nmat, int rows, int cols) {
    int total  = nmat * rows * cols;
    int stride = gridDim.x * blockDim.x;
    for (int i = blockIdx.x * blockDim.x + threadIdx.x; i < total; i += stride) {
        int rr = i % rows;
        int cc = (i / rows) % cols;
        int m  = i / (rows * cols);
        out[i] = (__bf16)in[(size_t)m * rows * cols + (size_t)rr * cols + cc];
    }
}

// ---------------------------------------------------------------------------
// QKV projection GEMM: xb[BT,C] (bf16) x wT[H,D,C] (bf16) -> Q/K row-major
// [B,H,T,D] bf16, V transposed [B,H,D,T] bf16.
// One wave computes a 16(row) x 64(D) tile. 4 waves / block.
// Weights are streamed into LDS by the Tensor Data Mover, double-buffered.
// ---------------------------------------------------------------------------
#define WTILE_B 4096   // 64 x 32 bf16 tile bytes

__global__ void __launch_bounds__(128)
qkv_gemm_kernel(const __bf16* __restrict__ xb,
                const __bf16* __restrict__ wqT,
                const __bf16* __restrict__ wkT,
                const __bf16* __restrict__ wvT,
                __bf16* __restrict__ Q,
                __bf16* __restrict__ K,
                __bf16* __restrict__ Vt) {
    __shared__ __align__(16) __bf16 wbuf[4][2][64][32];   // 32 KB, per-wave x2

    const int lane = threadIdx.x & 31;
    const int wib  = threadIdx.x >> 5;
    const int gw   = blockIdx.x * 4 + wib;          // global wave id

    const int mt   = gw % (BT_ / 16);               // row tile
    const int h    = (gw / (BT_ / 16)) % H_;        // head
    const int mat  = gw / ((BT_ / 16) * H_);        // 0=Q 1=K 2=V

    const int row0 = mt * 16;
    const int r    = lane & 15;
    const int hi   = (lane >> 4) & 1;
    const int ksel = hi * 8;

    const __bf16* wT   = (mat == 0) ? wqT : (mat == 1) ? wkT : wvT;
    const __bf16* Bw   = wT + (size_t)h * D_ * C_;  // [D, C]
    const __bf16* arow = xb + (size_t)(row0 + r) * C_;

    const unsigned lds_base = (unsigned)(wib * 2) * WTILE_B; // shared block @ LDS 0

    v8f acc[4];
    #pragma unroll
    for (int j = 0; j < 4; ++j) acc[j] = v8f_zero();

    // prime the TDM pipeline
    tdm_load_64x32(Bw, lds_base, C_ / 2);

    for (int kk = 0; kk < C_; kk += 32) {
        const int cur = (kk >> 5) & 1;
        // issue next tile (clamped redundant reload on the final iteration so
        // the wait below keeps a literal immediate)
        const int nxt = (kk + 32 < C_) ? (kk + 32) : (C_ - 32);
        tdm_load_64x32(Bw + nxt, lds_base + (unsigned)(cur ^ 1) * WTILE_B, C_ / 2);
        tdm_wait_1(&wbuf[wib][cur][0][0]);          // current tile resident

        __builtin_prefetch(arow + kk + 128, 0, 3);
        Frag16 a;
        a.u[0] = *(const uint4*)(arow + kk + ksel);
        a.u[1] = *(const uint4*)(arow + kk + 16 + ksel);
        #pragma unroll
        for (int j = 0; j < 4; ++j) {
            const __bf16* brow = &wbuf[wib][cur][16 * j + r][hi * 8];
            Frag16 bfr;
            bfr.u[0] = *(const uint4*)(brow);
            bfr.u[1] = *(const uint4*)(brow + 16);
            acc[j] = wmma_bf16(a.v, bfr.v, acc[j]);
        }
    }

    const int b  = row0 / T_;
    const int tl = row0 % T_;
    if (mat < 2) {
        // row-major [B,H,T,D]
        __bf16* O = (mat == 0 ? Q : K) + (((size_t)b * H_ + h) * T_ + tl) * D_;
        #pragma unroll
        for (int j = 0; j < 4; ++j) {
            int n = 16 * j + r;
            #pragma unroll
            for (int rr = 0; rr < 8; ++rr)
                O[(size_t)(rr + 8 * hi) * D_ + n] = (__bf16)acc[j][rr];
        }
    } else {
        // transposed [B,H,D,T]: 8 consecutive t values -> one b128 store
        __bf16* O = Vt + ((size_t)b * H_ + h) * (size_t)D_ * T_;
        #pragma unroll
        for (int j = 0; j < 4; ++j) {
            int n = 16 * j + r;
            union { __bf16 hv[8]; uint4 u; } pk;
            #pragma unroll
            for (int rr = 0; rr < 8; ++rr) pk.hv[rr] = (__bf16)acc[j][rr];
            *(uint4*)(O + (size_t)n * T_ + tl + 8 * hi) = pk.u;
        }
    }
}

// ---------------------------------------------------------------------------
// Flash attention: one wave per (b,h, 16-query-row tile), online softmax,
// key blocks of 32. ctx output is bf16 [B,T,C] with head-concat layout.
// ---------------------------------------------------------------------------
__global__ void __launch_bounds__(128)
attn_kernel(const __bf16* __restrict__ Q,
            const __bf16* __restrict__ K,
            const __bf16* __restrict__ Vt,
            __bf16* __restrict__ ctx) {
    __shared__ __align__(16) __bf16 pbuf[4][16][32];   // per-wave P staging

    const int lane = threadIdx.x & 31;
    const int wib  = threadIdx.x >> 5;
    const int gw   = blockIdx.x * 4 + wib;

    const int qt   = gw % (T_ / 16);
    const int bh   = gw / (T_ / 16);                   // 0 .. B*H-1
    const int b    = bh / H_;
    const int h    = bh % H_;

    const int row0 = qt * 16;
    const int r    = lane & 15;
    const int hi   = (lane >> 4) & 1;
    const int ksel = hi * 8;

    const __bf16* Qb = Q  + (size_t)bh * T_ * D_;
    const __bf16* Kb = K  + (size_t)bh * T_ * D_;
    const __bf16* Vb = Vt + (size_t)bh * D_ * T_;

    // Q A-fragments (16 rows x 64 d, two 16x32 fragments) -- loaded once
    Frag16 aq[2];
    {
        const __bf16* qrow = Qb + (size_t)(row0 + r) * D_;
        aq[0].u[0] = *(const uint4*)(qrow + 0  + ksel);
        aq[0].u[1] = *(const uint4*)(qrow + 16 + ksel);
        aq[1].u[0] = *(const uint4*)(qrow + 32 + ksel);
        aq[1].u[1] = *(const uint4*)(qrow + 48 + ksel);
    }

    v8f o[4];
    #pragma unroll
    for (int j = 0; j < 4; ++j) o[j] = v8f_zero();
    float mrow[8], lrow[8];
    #pragma unroll
    for (int rr = 0; rr < 8; ++rr) { mrow[rr] = -1e30f; lrow[rr] = 0.f; }

    const float scale  = 0.03125f;   // C^-0.5 = 1/32 (reference uses n_embd)
    const int   row_hi = row0 + 15;

    for (int s0 = 0; s0 <= row_hi; s0 += 32) {
        // --- scores: two 16x16 tiles, K-dim = D = 2 x 32 ---
        v8f sfr[2];
        sfr[0] = v8f_zero(); sfr[1] = v8f_zero();
        #pragma unroll
        for (int sc_i = 0; sc_i < 2; ++sc_i) {
            const int sc = s0 + 16 * sc_i;
            #pragma unroll
            for (int dd = 0; dd < 2; ++dd) {
                const __bf16* krow = Kb + (size_t)(sc + r) * D_ + dd * 32 + hi * 8;
                Frag16 bk;
                bk.u[0] = *(const uint4*)(krow);
                bk.u[1] = *(const uint4*)(krow + 16);
                sfr[sc_i] = wmma_bf16(aq[dd].v, bk.v, sfr[sc_i]);
            }
        }
        // --- scale + causal mask ---
        #pragma unroll
        for (int sc_i = 0; sc_i < 2; ++sc_i) {
            const int sg = s0 + 16 * sc_i + r;         // key index (fixed/lane)
            #pragma unroll
            for (int rr = 0; rr < 8; ++rr) {
                const int tg = row0 + rr + 8 * hi;     // query index
                float v = sfr[sc_i][rr] * scale;
                sfr[sc_i][rr] = (sg <= tg) ? v : -1e30f;
            }
        }
        // --- online softmax (reduce across the 16-lane group per row) ---
        #pragma unroll
        for (int rr = 0; rr < 8; ++rr) {
            float vmax = fmaxf(sfr[0][rr], sfr[1][rr]);
            #pragma unroll
            for (int off = 8; off >= 1; off >>= 1)
                vmax = fmaxf(vmax, __shfl_xor(vmax, off, 32));
            float mnew = fmaxf(mrow[rr], vmax);
            float corr = __expf(mrow[rr] - mnew);
            float p0   = __expf(sfr[0][rr] - mnew);
            float p1   = __expf(sfr[1][rr] - mnew);
            float ps   = p0 + p1;
            #pragma unroll
            for (int off = 8; off >= 1; off >>= 1)
                ps += __shfl_xor(ps, off, 32);
            lrow[rr] = lrow[rr] * corr + ps;
            mrow[rr] = mnew;
            sfr[0][rr] = p0; sfr[1][rr] = p1;
            o[0][rr] *= corr; o[1][rr] *= corr; o[2][rr] *= corr; o[3][rr] *= corr;
        }
        // --- re-layout P (C/D frag -> A frag) through LDS ---
        #pragma unroll
        for (int sc_i = 0; sc_i < 2; ++sc_i)
            #pragma unroll
            for (int rr = 0; rr < 8; ++rr)
                pbuf[wib][rr + 8 * hi][16 * sc_i + r] = (__bf16)sfr[sc_i][rr];
        asm volatile("s_wait_dscnt 0" ::: "memory");
        Frag16 ap;
        ap.u[0] = *(const uint4*)&pbuf[wib][r][ksel];
        ap.u[1] = *(const uint4*)&pbuf[wib][r][16 + ksel];
        // --- P (16x32) x V (32x64): Vt gives contiguous B-fragments ---
        #pragma unroll
        for (int j = 0; j < 4; ++j) {
            const __bf16* vrow = Vb + (size_t)(16 * j + r) * T_ + s0 + hi * 8;
            Frag16 bv;
            bv.u[0] = *(const uint4*)(vrow);
            bv.u[1] = *(const uint4*)(vrow + 16);
            o[j] = wmma_bf16(ap.v, bv.v, o[j]);
        }
    }

    // --- normalize and store head-concat ctx [B,T,C] ---
    #pragma unroll
    for (int rr = 0; rr < 8; ++rr) {
        const float inv = 1.0f / lrow[rr];
        const size_t rowoff = ((size_t)b * T_ + row0 + rr + 8 * hi) * C_ + h * D_;
        #pragma unroll
        for (int j = 0; j < 4; ++j)
            ctx[rowoff + 16 * j + r] = (__bf16)(o[j][rr] * inv);
    }
}

// ---------------------------------------------------------------------------
// Output projection: ctx[BT,C] (bf16) x wprojT[C,C] (bf16) + bias -> fp32,
// wproj tiles streamed through LDS by TDM (double-buffered).
// ---------------------------------------------------------------------------
__global__ void __launch_bounds__(128)
out_proj_kernel(const __bf16* __restrict__ ctx,
                const __bf16* __restrict__ wpT,
                const float*  __restrict__ bias,
                float* __restrict__ out) {
    __shared__ __align__(16) __bf16 wbuf[4][2][64][32];   // 32 KB

    const int lane = threadIdx.x & 31;
    const int wib  = threadIdx.x >> 5;
    const int gw   = blockIdx.x * 4 + wib;

    const int nt   = gw % (C_ / 64);     // 16 column groups of 64
    const int mt   = gw / (C_ / 64);     // 512 row tiles
    const int row0 = mt * 16;
    const int n0   = nt * 64;
    const int r    = lane & 15;
    const int hi   = (lane >> 4) & 1;
    const int ksel = hi * 8;

    const __bf16* arow = ctx + (size_t)(row0 + r) * C_;
    const __bf16* Bw   = wpT + (size_t)n0 * C_;   // 64 rows of wprojT

    const unsigned lds_base = (unsigned)(wib * 2) * WTILE_B;

    v8f acc[4];
    #pragma unroll
    for (int j = 0; j < 4; ++j) acc[j] = v8f_zero();

    tdm_load_64x32(Bw, lds_base, C_ / 2);

    for (int kk = 0; kk < C_; kk += 32) {
        const int cur = (kk >> 5) & 1;
        const int nxt = (kk + 32 < C_) ? (kk + 32) : (C_ - 32);
        tdm_load_64x32(Bw + nxt, lds_base + (unsigned)(cur ^ 1) * WTILE_B, C_ / 2);
        tdm_wait_1(&wbuf[wib][cur][0][0]);

        __builtin_prefetch(arow + kk + 128, 0, 3);
        Frag16 a;
        a.u[0] = *(const uint4*)(arow + kk + ksel);
        a.u[1] = *(const uint4*)(arow + kk + 16 + ksel);
        #pragma unroll
        for (int j = 0; j < 4; ++j) {
            const __bf16* brow = &wbuf[wib][cur][16 * j + r][hi * 8];
            Frag16 bfr;
            bfr.u[0] = *(const uint4*)(brow);
            bfr.u[1] = *(const uint4*)(brow + 16);
            acc[j] = wmma_bf16(a.v, bfr.v, acc[j]);
        }
    }

    #pragma unroll
    for (int j = 0; j < 4; ++j) {
        const int n  = n0 + 16 * j + r;
        const float bn = bias[n];
        #pragma unroll
        for (int rr = 0; rr < 8; ++rr)
            out[(size_t)(row0 + rr + 8 * hi) * C_ + n] = acc[j][rr] + bn;
    }
}

// ---------------------------------------------------------------------------
// Host-side launch
// ---------------------------------------------------------------------------
extern "C" void kernel_launch(void* const* d_in, const int* in_sizes, int n_in,
                              void* d_out, int out_size, void* d_ws, size_t ws_size,
                              hipStream_t stream) {
    const float* x     = (const float*)d_in[0];
    const float* wq    = (const float*)d_in[1];
    const float* wk    = (const float*)d_in[2];
    const float* wv    = (const float*)d_in[3];
    const float* wproj = (const float*)d_in[4];
    const float* bproj = (const float*)d_in[5];
    float*       out   = (float*)d_out;

    char* ws = (char*)d_ws;
    const size_t XB  = (size_t)BT_ * C_;          // 8,388,608
    const size_t WT  = (size_t)H_ * D_ * C_;      // 1,048,576
    const size_t WP  = (size_t)C_ * C_;           // 1,048,576
    const size_t QK  = (size_t)B_ * H_ * T_ * D_; // 8,388,608

    __bf16* xb    = (__bf16*)(ws);
    __bf16* wqT   = (__bf16*)(ws + 2 * XB);
    __bf16* wkT   = (__bf16*)(ws + 2 * (XB + WT));
    __bf16* wvT   = (__bf16*)(ws + 2 * (XB + 2 * WT));
    __bf16* wpT   = (__bf16*)(ws + 2 * (XB + 3 * WT));
    __bf16* Qb    = (__bf16*)(ws + 2 * (XB + 3 * WT + WP));
    __bf16* Kb    = (__bf16*)(ws + 2 * (XB + 3 * WT + WP + QK));
    __bf16* Vt    = (__bf16*)(ws + 2 * (XB + 3 * WT + WP + 2 * QK));
    __bf16* ctx   = (__bf16*)(ws + 2 * (XB + 3 * WT + WP + 3 * QK));

    // 1) convert inputs to bf16 (weights transposed for contiguous B-fragments)
    cvt_bf16_kernel<<<1024, 256, 0, stream>>>(x, xb, (int)XB);
    cvt_transpose_kernel<<<512, 256, 0, stream>>>(wq,    wqT, H_, C_, D_);
    cvt_transpose_kernel<<<512, 256, 0, stream>>>(wk,    wkT, H_, C_, D_);
    cvt_transpose_kernel<<<512, 256, 0, stream>>>(wv,    wvT, H_, C_, D_);
    cvt_transpose_kernel<<<512, 256, 0, stream>>>(wproj, wpT, 1,  C_, C_);

    // 2) QKV projection: 3 * 16 heads * 512 row-tiles = 24576 waves
    qkv_gemm_kernel<<<6144, 128, 0, stream>>>(xb, wqT, wkT, wvT, Qb, Kb, Vt);

    // 3) flash attention: 64 (b,h) * 128 row-tiles = 8192 waves
    attn_kernel<<<2048, 128, 0, stream>>>(Qb, Kb, Vt, ctx);

    // 4) output projection: 512 row-tiles * 16 col-groups = 8192 waves
    out_proj_kernel<<<2048, 128, 0, stream>>>(ctx, wpT, bproj, out);
}
